// MixtureOfExperts_75849122447476
// MI455X (gfx1250) — matrix operand order repaired and checked by reading, not compiled, not run on hardware
//
#include <hip/hip_runtime.h>
#include <cstdint>

#define T_TOKENS 8192
#define HID      1024
#define NE       8
#define BM       64
#define BN       128
#define BK       32
#define NK       (HID / BK)
#define LDA      (BK + 8)   // 40 ushorts = 80B rows: 16B-aligned, bank-spread
#define LDB      (BK + 8)

typedef __attribute__((ext_vector_type(16))) __bf16        v16bf;
typedef __attribute__((ext_vector_type(8)))  float         v8f;
typedef __attribute__((ext_vector_type(4)))  unsigned int  u32x4;
typedef __attribute__((ext_vector_type(4)))  float         f32x4;

__device__ __forceinline__ unsigned short moe_f2bf(float f) {
    unsigned u = __float_as_uint(f);
    u += 0x7fffu + ((u >> 16) & 1u);          // round-to-nearest-even
    return (unsigned short)(u >> 16);
}
__device__ __forceinline__ unsigned moe_pk2(float a, float b) {
    return (unsigned)moe_f2bf(a) | ((unsigned)moe_f2bf(b) << 16);
}
__device__ __forceinline__ v16bf moe_mk16(u32x4 lo, u32x4 hi) {
    union { struct { u32x4 a, b; } s; v16bf v; } u;
    u.s.a = lo; u.s.b = hi;
    return u.v;
}

// CDNA5 async DMA: 16B from global -> LDS, per-lane addresses, tracked by ASYNCcnt.
__device__ __forceinline__ void moe_async_b128(unsigned lds_off, const void* gaddr) {
    asm volatile("global_load_async_to_lds_b128 %0, %1, off"
                 :: "v"(lds_off), "v"(gaddr) : "memory");
}
template <int N>
__device__ __forceinline__ void moe_wait_async() {
#if __has_builtin(__builtin_amdgcn_s_wait_asynccnt)
    __builtin_amdgcn_s_wait_asynccnt(N);
#else
    asm volatile("s_wait_asynccnt %0" :: "i"(N) : "memory");
#endif
}

// ---------------- routing ----------------
__global__ void moe_zero(int* counts) {
    if (threadIdx.x < NE) counts[threadIdx.x] = 0;
}

__global__ void moe_route(const float* __restrict__ gate, float* __restrict__ scl,
                          int* __restrict__ eo, int* __restrict__ counts) {
    int t = blockIdx.x * blockDim.x + threadIdx.x;
    if (t >= T_TOKENS) return;
    const float* g = gate + (size_t)t * NE;
    float m = g[0]; int e = 0;
    #pragma unroll
    for (int i = 1; i < NE; ++i) { float v = g[i]; if (v > m) { m = v; e = i; } }
    float s = 0.f;
    #pragma unroll
    for (int i = 0; i < NE; ++i) s += __expf(g[i] - m);
    scl[t] = 1.0f / s;          // softmax value at the argmax
    eo[t]  = e;
    atomicAdd(&counts[e], 1);
}

__global__ void moe_scan(const int* __restrict__ counts, int* __restrict__ base,
                         int* __restrict__ cursor) {
    if (threadIdx.x == 0) {
        int acc = 0;
        for (int e = 0; e < NE; ++e) { base[e] = acc; cursor[e] = acc; acc += counts[e]; }
    }
}

__global__ void moe_scatter(const int* __restrict__ eo, int* __restrict__ cursor,
                            int* __restrict__ perm) {
    int t = blockIdx.x * blockDim.x + threadIdx.x;
    if (t >= T_TOKENS) return;
    int pos = atomicAdd(&cursor[eo[t]], 1);
    perm[pos] = t;
}

// ---------------- X f32 -> bf16 (keeps conversion out of the GEMM hot loop) ----------------
__global__ __launch_bounds__(256) void moe_xb(const float* __restrict__ X,
                                              unsigned short* __restrict__ Xb) {
    size_t i = ((size_t)blockIdx.x * 256 + threadIdx.x) * 8;
    const f32x4* p = (const f32x4*)(X + i);
    f32x4 f0 = p[0], f1 = p[1];
    u32x4 v;
    v.x = moe_pk2(f0.x, f0.y); v.y = moe_pk2(f0.z, f0.w);
    v.z = moe_pk2(f1.x, f1.y); v.w = moe_pk2(f1.z, f1.w);
    *(u32x4*)(Xb + i) = v;
}

// ---------------- W f32 -> bf16, transposed to [e][n][k] ----------------
__global__ __launch_bounds__(256) void moe_wt(const float* __restrict__ W,
                                              unsigned short* __restrict__ Wt) {
    __shared__ float tile[32][33];
    const int e  = blockIdx.z;
    const int n0 = blockIdx.x * 32;
    const int k0 = blockIdx.y * 32;
    const int tx = threadIdx.x;   // 0..31
    const int ty = threadIdx.y;   // 0..7
    const float* Wp = W + (size_t)e * HID * HID;
    #pragma unroll
    for (int r = ty; r < 32; r += 8)
        tile[r][tx] = Wp[(size_t)(k0 + r) * HID + n0 + tx];
    __syncthreads();
    unsigned short* Wo = Wt + (size_t)e * HID * HID;
    #pragma unroll
    for (int r = ty; r < 32; r += 8)
        Wo[(size_t)(n0 + r) * HID + k0 + tx] = moe_f2bf(tile[tx][r]);
}

// ---------------- grouped GEMM: out[t] = scale[t] * (x[t] @ W[e] + b[e]) ----------------
__global__ __launch_bounds__(256) void moe_gemm(
    const unsigned short* __restrict__ Xb, const unsigned short* __restrict__ Wt,
    const float* __restrict__ bias, const int* __restrict__ counts,
    const int* __restrict__ base, const int* __restrict__ perm,
    const float* __restrict__ scl, float* __restrict__ out)
{
    const int bx  = blockIdx.x;
    const int e   = bx >> 7;           // 128 = T_TOKENS / BM max M-tiles per expert
    const int mt  = bx & 127;
    const int cnt = counts[e];
    const int m0  = mt * BM;
    if (m0 >= cnt) return;
    const int bas = base[e];
    const int n0  = blockIdx.y * BN;

    const int tid  = threadIdx.x;
    const int lane = tid & 31;
    const int wave = tid >> 5;
    const int wm   = wave & 1;         // wave row (2): rows wm*32..+31
    const int wn   = wave >> 1;        // wave col (4): cols wn*32..+31

    __shared__ __align__(16) unsigned short As[2][BM][LDA];
    __shared__ __align__(16) unsigned short Bs[2][BN][LDB];
    __shared__ int   tokS[BM];
    __shared__ float sclS[BM];

    if (tid < BM) {
        int gm = m0 + tid;
        int t = -1; float s = 0.f;
        if (gm < cnt) { t = perm[bas + gm]; s = scl[t]; }
        tokS[tid] = t; sclS[tid] = s;
    }
    __syncthreads();

    const int arow = tid >> 2, aseg = tid & 3;   // A stage: 64 rows x 4 segs of 8 bf16
    const int brow = tid >> 1, bhalf = tid & 1;  // B stage: 128 rows x 2 halves of 16 bf16
    const int atok = tokS[arow];
    // Pad rows read Xb row 0 (valid memory; their C rows are never stored).
    const unsigned short* aBase = Xb + (atok >= 0 ? (size_t)atok * HID : 0) + aseg * 8;
    const unsigned short* bBase = Wt + (((size_t)(e << 10) + n0 + brow) << 10) + bhalf * 16;
    const unsigned aLds  = (unsigned)(uintptr_t)&As[0][arow][aseg * 8];
    const unsigned bLds  = (unsigned)(uintptr_t)&Bs[0][brow][bhalf * 16];
    const unsigned aBufB = (unsigned)sizeof(As[0]);
    const unsigned bBufB = (unsigned)sizeof(Bs[0]);

    // issue one K-step's tiles (3 async b128 per thread) into buffer `buf`
    auto issue = [&](int k0, int buf) {
        moe_async_b128(aLds + buf * aBufB, aBase + k0);
        moe_async_b128(bLds + buf * bBufB, bBase + k0);
        moe_async_b128(bLds + buf * bBufB + 16, bBase + k0 + 8);
    };

    v8f c00 = {}, c01 = {}, c10 = {}, c11 = {};

    issue(0, 0);
    for (int i = 0; i < NK; ++i) {
        const int cur = i & 1;
        if (i + 1 < NK) {
            issue((i + 1) * BK, cur ^ 1);   // prefetch next buffer (overlaps compute)
            moe_wait_async<3>();            // current buffer's 3 ops complete
        } else {
            moe_wait_async<0>();
        }
        __syncthreads();                    // current tiles visible to all waves

        const int g = lane >> 4, lr = lane & 15;
        // A frags: lane(row M=lr, group g) needs K in [g*8, g*8+8) and [16+g*8, ...)
        const int r0 = wm * 32 + lr;
        v16bf a0 = moe_mk16(*(const u32x4*)&As[cur][r0][g * 8],
                            *(const u32x4*)&As[cur][r0][16 + g * 8]);
        v16bf a1 = moe_mk16(*(const u32x4*)&As[cur][r0 + 16][g * 8],
                            *(const u32x4*)&As[cur][r0 + 16][16 + g * 8]);
        // B frags: lane(col N=lr, group g) needs K in [g*8, g*8+8) and [16+g*8, ...)
        const int nb = wn * 32 + lr;
        v16bf b0 = moe_mk16(*(const u32x4*)&Bs[cur][nb][g * 8],
                            *(const u32x4*)&Bs[cur][nb][16 + g * 8]);
        v16bf b1 = moe_mk16(*(const u32x4*)&Bs[cur][nb + 16][g * 8],
                            *(const u32x4*)&Bs[cur][nb + 16][16 + g * 8]);

        c00 = __builtin_amdgcn_wmma_f32_16x16x32_bf16(false, a0, false, b0, (short)0, c00, false, false);
        c01 = __builtin_amdgcn_wmma_f32_16x16x32_bf16(false, a0, false, b1, (short)0, c01, false, false);
        c10 = __builtin_amdgcn_wmma_f32_16x16x32_bf16(false, a1, false, b0, (short)0, c10, false, false);
        c11 = __builtin_amdgcn_wmma_f32_16x16x32_bf16(false, a1, false, b1, (short)0, c11, false, false);
        __syncthreads();                    // readers done before buffer is re-filled
    }

    // ---- epilogue: out[t][n] = scale[t] * (acc + b[e][n]) ----
    const int g = lane >> 4, lr = lane & 15;
    const float* be = bias + (size_t)e * HID;
    #pragma unroll
    for (int q = 0; q < 4; ++q) {
        const v8f& c = (q == 0) ? c00 : (q == 1) ? c01 : (q == 2) ? c10 : c11;
        const int mi = q >> 1, ni = q & 1;
        const int nG = n0 + wn * 32 + ni * 16 + lr;
        const float bv = be[nG];
        #pragma unroll
        for (int j = 0; j < 8; ++j) {
            int mrow = wm * 32 + mi * 16 + g * 8 + j;   // C layout: VGPR j, lane-group g
            int t = tokS[mrow];
            if (t >= 0)
                out[(size_t)t * HID + nG] = sclS[mrow] * (c[j] + bv);
        }
    }
}

extern "C" void kernel_launch(void* const* d_in, const int* in_sizes, int n_in,
                              void* d_out, int out_size, void* d_ws, size_t ws_size,
                              hipStream_t stream) {
    (void)in_sizes; (void)n_in; (void)out_size; (void)ws_size;
    const float* inp  = (const float*)d_in[0];
    const float* gate = (const float*)d_in[1];
    const float* W    = (const float*)d_in[2];
    const float* bias = (const float*)d_in[3];
    float* out = (float*)d_out;

    // workspace: Wt bf16 (16MB) | Xb bf16 (16MB) | scale | expert_of | perm | counts | base | cursor
    unsigned short* Wt = (unsigned short*)d_ws;
    unsigned short* Xb = Wt + (size_t)NE * HID * HID;
    float* scl  = (float*)(Xb + (size_t)T_TOKENS * HID);
    int* eo     = (int*)(scl + T_TOKENS);
    int* perm   = eo + T_TOKENS;
    int* counts = perm + T_TOKENS;
    int* base   = counts + NE;
    int* cursor = base + NE;

    moe_zero<<<1, 32, 0, stream>>>(counts);
    moe_route<<<T_TOKENS / 256, 256, 0, stream>>>(gate, scl, eo, counts);
    moe_scan<<<1, 32, 0, stream>>>(counts, base, cursor);
    moe_scatter<<<T_TOKENS / 256, 256, 0, stream>>>(eo, cursor, perm);
    moe_xb<<<(T_TOKENS * HID / 8) / 256, 256, 0, stream>>>(inp, Xb);
    moe_wt<<<dim3(HID / 32, HID / 32, NE), dim3(32, 8), 0, stream>>>(W, Wt);
    moe_gemm<<<dim3(NE * (T_TOKENS / BM), HID / BN), 256, 0, stream>>>(
        Xb, Wt, bias, counts, base, perm, scl, out);
}
